// MultiHeadAttention_tr2_74509092651465
// MI455X (gfx1250) — matrix operand order repaired and checked by reading, not compiled, not run on hardware
//
#include <hip/hip_runtime.h>
#include <hip/hip_bf16.h>

// ---------------- types ----------------
typedef __attribute__((ext_vector_type(16))) __bf16          bf16x16;
typedef __attribute__((ext_vector_type(8)))  float           f32x8;
typedef __attribute__((ext_vector_type(4)))  float           f32x4;
typedef __attribute__((ext_vector_type(8)))  unsigned short  us8;
typedef __attribute__((ext_vector_type(4)))  unsigned short  us4;
typedef int v4i __attribute__((vector_size(4 * sizeof(int))));

union Frag {
    bf16x16 b;
    us8     h[2];
};

__device__ __forceinline__ unsigned short f2bf(float f) {
    unsigned int u = __float_as_uint(f);
    u += 0x7FFFu + ((u >> 16) & 1u);   // round to nearest even
    return (unsigned short)(u >> 16);
}

// ---------------- async global->LDS copy (16 bytes / lane) ----------------
#if defined(__gfx1250__) && __has_builtin(__builtin_amdgcn_global_load_async_to_lds_b128)
#define USE_ASYNC 1
#else
#define USE_ASYNC 0
#endif

__device__ __forceinline__ void cp16(const unsigned short* g, unsigned short* l) {
#if USE_ASYNC
    __builtin_amdgcn_global_load_async_to_lds_b128(
        (__attribute__((address_space(1))) v4i*)(void*)g,
        (__attribute__((address_space(3))) v4i*)(void*)l, 0, 0);
#else
    *(us8*)l = *(const us8*)g;
#endif
}

__device__ __forceinline__ void async_wait_all() {
#if USE_ASYNC
#if __has_builtin(__builtin_amdgcn_s_wait_asynccnt)
    __builtin_amdgcn_s_wait_asynccnt(0);
#else
    asm volatile("s_wait_asynccnt 0x0" ::: "memory");
#endif
#endif
}

// ---------------- WMMA GEMM (both operands K-contiguous bf16) ----------------
// C[M,N] = A[M,K] * Bt[N,K]^T, tile 128x128, BK=32, 256 thr = 8 waves,
// each wave 32(M) x 64(N) => 2x4 tiles of v_wmma_f32_16x16x32_bf16.
// Double-buffered LDS, tiles filled with GLOBAL_LOAD_ASYNC_TO_LDS_B128.
//
// EM: 0 = bf16 out, +bias        (qp, kp projections)
//     1 = f32 out, *0.03125 + mask*(-1e9)   (logits)
//     2 = bf16 out               (ctx)
//     3 = f32 out, +bias         (final)
//     4 = bf16 out TRANSPOSED, +bias   (vp^T projection; C[col*ldc + row])
template <int EM>
__global__ __launch_bounds__(256) void gemm_wmma(
    const unsigned short* __restrict__ A, const unsigned short* __restrict__ Bt,
    const float* __restrict__ bias, const float* __restrict__ mask,
    void* __restrict__ Cp,
    int lda, int ldbt, int ldc, int K,
    size_t aBS, size_t bBS, size_t cBS, size_t mBS)
{
    __shared__ __attribute__((aligned(16))) unsigned short sA[2][128 * 32]; // 2 x 8 KB
    __shared__ __attribute__((aligned(16))) unsigned short sB[2][128 * 32]; // 2 x 8 KB

    const int tid    = threadIdx.x;
    const int lane   = tid & 31;
    const int wave   = tid >> 5;
    const int l16    = lane & 15;
    const int hi     = lane >> 4;
    const int m0     = (wave >> 1) * 32;
    const int n0     = (wave & 1) * 64;
    const int blockM = blockIdx.y * 128;
    const int blockN = blockIdx.x * 128;
    const size_t z   = blockIdx.z;

    const unsigned short* Ab = A  + z * aBS;
    const unsigned short* Bb = Bt + z * bBS;

    // per-thread tile-load: 2 x 16B for A, 2 x 16B for B
    auto load_tile = [&](int kt, int buf) {
        #pragma unroll
        for (int i = 0; i < 2; ++i) {
            int idx = tid + i * 256;       // 0..511
            int row = idx >> 2;            // 0..127
            int c8  = idx & 3;             // 0..3
            cp16(Ab + (size_t)(blockM + row) * lda  + kt * 32 + c8 * 8,
                 &sA[buf][row * 32 + c8 * 8]);
            cp16(Bb + (size_t)(blockN + row) * ldbt + kt * 32 + c8 * 8,
                 &sB[buf][row * 32 + c8 * 8]);
        }
    };

    f32x8 acc[2][4] = {};

    const int ksteps = K >> 5;
    load_tile(0, 0);

    for (int kt = 0; kt < ksteps; ++kt) {
        const int cur = kt & 1;
        async_wait_all();
        __syncthreads();
        if (kt + 1 < ksteps) load_tile(kt + 1, cur ^ 1);

        // fragments per ISA 7.12.2 layouts
        Frag a[2], bb[4];
        #pragma unroll
        for (int mt = 0; mt < 2; ++mt) {
            int r = m0 + mt * 16 + l16;
            a[mt].h[0] = *(const us8*)&sA[cur][r * 32 + hi * 8];       // K 0..7 / 8..15
            a[mt].h[1] = *(const us8*)&sA[cur][r * 32 + 16 + hi * 8];  // K 16..23 / 24..31
        }
        #pragma unroll
        for (int nt = 0; nt < 4; ++nt) {
            int c = n0 + nt * 16 + l16;
            bb[nt].h[0] = *(const us8*)&sB[cur][c * 32 + hi * 8];      // K 0..7 / 8..15
            bb[nt].h[1] = *(const us8*)&sB[cur][c * 32 + 16 + hi * 8]; // K 16..23 / 24..31
        }

        #pragma unroll
        for (int mt = 0; mt < 2; ++mt)
            #pragma unroll
            for (int nt = 0; nt < 4; ++nt)
                acc[mt][nt] = __builtin_amdgcn_wmma_f32_16x16x32_bf16(
                    false, a[mt].b, false, bb[nt].b, (short)0, acc[mt][nt], false, false);
    }

    // ---- epilogue ----
    const size_t cBase = z * cBS;
    #pragma unroll
    for (int mt = 0; mt < 2; ++mt) {
        #pragma unroll
        for (int nt = 0; nt < 4; ++nt) {
            const int col = blockN + n0 + nt * 16 + l16;
            #pragma unroll
            for (int r = 0; r < 8; ++r) {
                const int row = blockM + m0 + mt * 16 + hi * 8 + r;
                float v = acc[mt][nt][r];
                if (EM == 0) {
                    v += bias[col];
                    ((unsigned short*)Cp)[cBase + (size_t)row * ldc + col] = f2bf(v);
                } else if (EM == 1) {
                    v = v * 0.03125f + mask[z * mBS + col] * (-1.0e9f);
                    ((float*)Cp)[cBase + (size_t)row * ldc + col] = v;
                } else if (EM == 2) {
                    ((unsigned short*)Cp)[cBase + (size_t)row * ldc + col] = f2bf(v);
                } else if (EM == 3) {
                    v += bias[col];
                    ((float*)Cp)[cBase + (size_t)row * ldc + col] = v;
                } else { // EM == 4 : transposed bf16 store
                    v += bias[col];
                    ((unsigned short*)Cp)[cBase + (size_t)col * ldc + row] = f2bf(v);
                }
            }
        }
    }
}

// ---------------- prepass: f32 -> bf16 straight copy ----------------
__global__ __launch_bounds__(256) void f32_to_bf16(
    const float* __restrict__ in, unsigned short* __restrict__ out)
{
    const size_t i = ((size_t)blockIdx.x * 256 + threadIdx.x) * 8;
    f32x4 a = *(const f32x4*)(in + i);
    f32x4 b = *(const f32x4*)(in + i + 4);
    us4 ha, hb;
    ha.x = f2bf(a.x); ha.y = f2bf(a.y); ha.z = f2bf(a.z); ha.w = f2bf(a.w);
    hb.x = f2bf(b.x); hb.y = f2bf(b.y); hb.z = f2bf(b.z); hb.w = f2bf(b.w);
    *(us4*)(out + i)     = ha;
    *(us4*)(out + i + 4) = hb;
}

// ---------------- prepass: f32 [K x N] -> bf16 [N x K] (weights) ----------------
__global__ __launch_bounds__(256) void transpose_f32_to_bf16(
    const float* __restrict__ in, unsigned short* __restrict__ out, int N, int K)
{
    __shared__ unsigned short t[32][33];
    const int tx = threadIdx.x & 31;
    const int ty = threadIdx.x >> 5;      // 0..7
    const int k0 = blockIdx.y * 32, n0 = blockIdx.x * 32;
    #pragma unroll
    for (int i = 0; i < 4; ++i) {
        int r = ty + i * 8;               // k within tile
        t[tx][r] = f2bf(in[(size_t)(k0 + r) * N + n0 + tx]);
    }
    __syncthreads();
    #pragma unroll
    for (int i = 0; i < 4; ++i) {
        int r = ty + i * 8;               // n within tile
        out[(size_t)(n0 + r) * K + k0 + tx] = t[r][tx];
    }
}

// ---------------- row softmax: 2048 cols, one block per row ----------------
__global__ __launch_bounds__(256) void softmax_kernel(
    const float* __restrict__ logits, unsigned short* __restrict__ attn)
{
    __shared__ float red[256];
    const int tid = threadIdx.x;
    const size_t row = blockIdx.x;
    const float* in = logits + row * 2048;

    float v[8];
    float m = -3.4e38f;
    #pragma unroll
    for (int i = 0; i < 8; ++i) {
        v[i] = in[tid + i * 256];
        m = fmaxf(m, v[i]);
    }
    red[tid] = m; __syncthreads();
    for (int s = 128; s > 0; s >>= 1) {
        if (tid < s) red[tid] = fmaxf(red[tid], red[tid + s]);
        __syncthreads();
    }
    m = red[0]; __syncthreads();

    float sum = 0.0f;
    #pragma unroll
    for (int i = 0; i < 8; ++i) {
        v[i] = __expf(v[i] - m);
        sum += v[i];
    }
    red[tid] = sum; __syncthreads();
    for (int s = 128; s > 0; s >>= 1) {
        if (tid < s) red[tid] += red[tid + s];
        __syncthreads();
    }
    const float inv = 1.0f / red[0];

    unsigned short* out = attn + row * 2048;
    #pragma unroll
    for (int i = 0; i < 8; ++i)
        out[tid + i * 256] = f2bf(v[i] * inv);
}

// ---------------- launch ----------------
extern "C" void kernel_launch(void* const* d_in, const int* in_sizes, int n_in,
                              void* d_out, int out_size, void* d_ws, size_t ws_size,
                              hipStream_t stream)
{
    (void)in_sizes; (void)n_in; (void)out_size; (void)ws_size;

    constexpr int    B = 4, S = 2048, D = 1024, E = 1024;
    constexpr size_t MB = 1024ull * 1024ull;

    // input order: v, k, q, mask, wq_w, wq_b, wk_w, wk_b, wv_w, wv_b, out_w, out_b
    const float* v_in  = (const float*)d_in[0];
    const float* k_in  = (const float*)d_in[1];
    const float* q_in  = (const float*)d_in[2];
    const float* maskp = (const float*)d_in[3];
    const float* wq_w  = (const float*)d_in[4];
    const float* wq_b  = (const float*)d_in[5];
    const float* wk_w  = (const float*)d_in[6];
    const float* wk_b  = (const float*)d_in[7];
    const float* wv_w  = (const float*)d_in[8];
    const float* wv_b  = (const float*)d_in[9];
    const float* out_w = (const float*)d_in[10];
    const float* out_b = (const float*)d_in[11];
    float* out = (float*)d_out;

    // workspace layout (byte offsets), attn aliases dead qb/kb; total 184 MB
    char* ws = (char*)d_ws;
    unsigned short* qp     = (unsigned short*)(ws + 0 * MB);    // 16 MB
    unsigned short* kp     = (unsigned short*)(ws + 16 * MB);   // 16 MB
    unsigned short* vpT    = (unsigned short*)(ws + 32 * MB);   // 16 MB  [B][E][S]
    unsigned short* ctx    = (unsigned short*)(ws + 48 * MB);   // 16 MB
    float*          logits = (float*)         (ws + 64 * MB);   // 64 MB
    unsigned short* attn   = (unsigned short*)(ws + 128 * MB);  // 32 MB (aliases qb,kb)
    unsigned short* qb     = (unsigned short*)(ws + 128 * MB);  // 16 MB (dead before attn)
    unsigned short* kb     = (unsigned short*)(ws + 144 * MB);  // 16 MB (dead before attn)
    unsigned short* vb     = (unsigned short*)(ws + 160 * MB);  // 16 MB
    unsigned short* wqT    = (unsigned short*)(ws + 176 * MB);  // 2 MB  [E][D]
    unsigned short* wkT    = (unsigned short*)(ws + 178 * MB);  // 2 MB
    unsigned short* wvT    = (unsigned short*)(ws + 180 * MB);  // 2 MB
    unsigned short* outT   = (unsigned short*)(ws + 182 * MB);  // 2 MB  [D][E]

    const dim3 blk(256);

    // 0) prepass: bf16 copies of activations, transposed bf16 weights
    {
        dim3 gc((size_t)B * S * D / 2048, 1, 1);    // 8 elems / thread
        f32_to_bf16<<<gc, blk, 0, stream>>>(q_in, qb);
        f32_to_bf16<<<gc, blk, 0, stream>>>(k_in, kb);
        f32_to_bf16<<<gc, blk, 0, stream>>>(v_in, vb);
        dim3 gt(E / 32, D / 32, 1);
        transpose_f32_to_bf16<<<gt, blk, 0, stream>>>(wq_w, wqT, E, D);
        transpose_f32_to_bf16<<<gt, blk, 0, stream>>>(wk_w, wkT, E, D);
        transpose_f32_to_bf16<<<gt, blk, 0, stream>>>(wv_w, wvT, E, D);
        dim3 go(D / 32, E / 32, 1);
        transpose_f32_to_bf16<<<go, blk, 0, stream>>>(out_w, outT, D, E);
    }

    // 1) projections (per batch): qp,kp normal; vp stored transposed [E][S]
    {
        dim3 g(E / 128, S / 128, B);   // (8, 16, 4)
        gemm_wmma<0><<<g, blk, 0, stream>>>(qb, wqT, wq_b, nullptr, qp,
                                            D, D, E, D,
                                            (size_t)S * D, 0, (size_t)S * E, 0);
        gemm_wmma<0><<<g, blk, 0, stream>>>(kb, wkT, wk_b, nullptr, kp,
                                            D, D, E, D,
                                            (size_t)S * D, 0, (size_t)S * E, 0);
        gemm_wmma<4><<<g, blk, 0, stream>>>(vb, wvT, wv_b, nullptr, vpT,
                                            D, D, S, D,
                                            (size_t)S * D, 0, (size_t)E * S, 0);
    }

    // 2) logits = (qp @ kp^T)/32 + mask*(-1e9)  (kp rows are already K-contiguous)
    {
        dim3 g(S / 128, S / 128, B);   // (16, 16, 4)
        gemm_wmma<1><<<g, blk, 0, stream>>>(qp, kp, nullptr, maskp, logits,
                                            E, E, S, E,
                                            (size_t)S * E, (size_t)S * E,
                                            (size_t)S * S, (size_t)S);
    }

    // 3) softmax rows -> bf16 attn
    softmax_kernel<<<dim3(B * S), blk, 0, stream>>>(logits, attn);

    // 4) ctx = attn @ vp  (vp^T rows are K-contiguous)
    {
        dim3 g(E / 128, S / 128, B);   // (8, 16, 4)
        gemm_wmma<2><<<g, blk, 0, stream>>>(attn, vpT, nullptr, nullptr, ctx,
                                            S, S, E, S,
                                            (size_t)S * S, (size_t)E * S,
                                            (size_t)S * E, 0);
    }

    // 5) out = ctx @ out_w + out_b  (flatten batches)
    {
        dim3 g(D / 128, (B * S) / 128, 1);   // (8, 64, 1)
        gemm_wmma<3><<<g, blk, 0, stream>>>(ctx, outT, out_b, nullptr, out,
                                            E, E, D, E,
                                            0, 0, 0, 0);
    }
}
